// GlobalPerceiver_22333829940019
// MI455X (gfx1250) — matrix operand order repaired and checked by reading, not compiled, not run on hardware
//
#include <hip/hip_runtime.h>
#include <hip/hip_bf16.h>
#include <math.h>

// ---- types for WMMA ----
typedef __attribute__((ext_vector_type(16))) __bf16 v16bf;
typedef __attribute__((ext_vector_type(8)))  __bf16 v8bf;
typedef __attribute__((ext_vector_type(8)))  float  v8f;

#define B_    64
#define N_    4096
#define D_    240
#define H_    8
#define HD_   30
#define FF_   1024
#define CH_   128        // kv rows per chunk (2 row-blocks per wave)
#define NCH_  (N_/CH_)   // 32 chunks
#define KSTR  241        // padded LDS row stride (odd -> no bank conflicts)

// ---- gfx1250 async global->LDS load (tracked by ASYNCcnt) ----
__device__ __forceinline__ void async_b128_to_lds(void* lds_dst, const void* gsrc) {
  unsigned lo = (unsigned)(unsigned long long)lds_dst;  // LDS offset = addr[31:0]
  asm volatile("global_load_async_to_lds_b128 %0, %1, off"
               :: "v"(lo), "v"(gsrc) : "memory");
}
__device__ __forceinline__ void wait_asynccnt0() {
  asm volatile("s_wait_asynccnt 0x0" ::: "memory");
}

// ---------------- init x = [state ; learnable+modality] ----------------
__global__ void init_x_kernel(const float* __restrict__ state,
                              const float* __restrict__ learn,
                              const float* __restrict__ mod,
                              float* __restrict__ x) {
  int idx = blockIdx.x * 256 + threadIdx.x;
  if (idx >= B_ * 3 * D_) return;
  int b = idx / (3 * D_), r = idx % (3 * D_), t = r / D_, d = r % D_;
  float v = (t == 0) ? state[b * D_ + d] : (learn[(t - 1) * D_ + d] + mod[d]);
  x[idx] = v;
}

// ---------------- pack Wk/Wv -> bf16, cols padded 240->256 ----------------
// WP layout: [layer][0=K,1=V][240][256] bf16
__global__ void pack_w_kernel(const float* __restrict__ Wk,
                              const float* __restrict__ Wv,
                              __bf16* __restrict__ WP) {
  int idx = blockIdx.x * 256 + threadIdx.x;        // 2*2*240*256 = 245760
  if (idx >= 2 * 2 * 240 * 256) return;
  int c = idx & 255;
  int rowidx = idx >> 8;                            // 0..959
  int d = rowidx % 240;
  int ls = rowidx / 240;                            // 0..3
  int l = ls >> 1, s = ls & 1;
  float v = 0.f;
  if (c < 240) {
    const float* W = s ? Wv : Wk;
    v = W[((size_t)l * 240 + d) * 240 + c];
  }
  WP[idx] = (__bf16)v;
}

// ---------------- pack kv -> bf16, cols padded 240->256 ----------------
// kvP layout: [b][n][256] bf16 (134 MB -> fits MI455X 192MB L2)
__global__ void pack_kv_kernel(const float* __restrict__ kv,
                               __bf16* __restrict__ kvP) {
  size_t t = (size_t)blockIdx.x * 256 + threadIdx.x;   // one thread per 8 cols
  size_t bn = t >> 5;
  int c8 = (int)(t & 31);
  if (bn >= (size_t)B_ * N_) return;
  __bf16* dst = kvP + bn * 256 + c8 * 8;
  if (c8 < 30) {
    const float* src = kv + bn * 240 + c8 * 8;
#pragma unroll
    for (int e = 0; e < 8; ++e) dst[e] = (__bf16)src[e];
  } else {
#pragma unroll
    for (int e = 0; e < 8; ++e) dst[e] = (__bf16)0.f;
  }
}

// ---------------- q projection + RoPE (folds 1/sqrt(HD)) ----------------
__global__ __launch_bounds__(256)
void qproj_rope_kernel(const float* __restrict__ x,
                       const float* __restrict__ Wq, const float* __restrict__ bq,
                       const float* __restrict__ hand, const float* __restrict__ head,
                       float* __restrict__ qrot, int layer) {
  __shared__ float xs[720];
  __shared__ float qp[720];
  int b = blockIdx.x, tid = threadIdx.x;
  for (int i = tid; i < 720; i += 256) xs[i] = x[b * 720 + i];
  __syncthreads();
  for (int i = tid; i < 720; i += 256) {
    int t = i / 240, d = i % 240;
    float acc = bq[layer * 240 + d];
    const float* wrow = Wq + ((size_t)layer * 240 + d) * 240;
    for (int c = 0; c < 240; ++c) acc += xs[t * 240 + c] * wrow[c];
    qp[i] = acc;
  }
  __syncthreads();
  const float scale = 0.18257418583505536f;  // 1/sqrt(30)
  for (int i = tid; i < 360; i += 256) {     // 3 tokens * 8 heads * 15 pairs
    int t = i / 120, rem = i % 120, h = rem / 15, p = rem % 15;
    float coord = 0.f;
    if (t == 1) coord = hand[b * 3 + p / 5];
    else if (t == 2) coord = head[b * 3 + p / 5];
    float invf = __expf(-(float)(p % 5) * 0.2f * 9.21034037197618f); // 10000^(-2m/10)
    float sn, cs;
    __sincosf(coord * invf, &sn, &cs);
    int c0 = t * 240 + h * 30 + 2 * p;
    float x1 = qp[c0], x2 = qp[c0 + 1];
    qrot[b * 720 + c0]     = (x1 * cs - x2 * sn) * scale;
    qrot[b * 720 + c0 + 1] = (x1 * sn + x2 * cs) * scale;
  }
}

// ---- fused K/V projection (WMMA bf16, 2 row-blocks/wave) + RoPE(K) + flash partial ----
// grid = B*NCH_ blocks, 256 threads (8 wave32)
__global__ __launch_bounds__(256)
void attn_partial_kernel(const __bf16* __restrict__ kvP,
                         const __bf16* __restrict__ WP,
                         const float* __restrict__ bk, const float* __restrict__ bv,
                         const float* __restrict__ qrot,
                         const float* __restrict__ coords,
                         float* __restrict__ part, int layer) {
  __shared__ __align__(16) float Kl[CH_ * KSTR];
  __shared__ __align__(16) float Vl[CH_ * KSTR];
  __shared__ __align__(16) float qs[720];
  __shared__ __align__(16) float cs[CH_ * 3];
  int b = blockIdx.x >> 5;
  int chunk = blockIdx.x & 31;
  int tid = threadIdx.x;
  int w = tid >> 5, ln = tid & 31;
  int g = ln >> 4, lr = ln & 15;

  // ---- async-stage q vectors (180 x b128) and chunk coords (96 x b128) ----
  if (tid < 180) async_b128_to_lds(&qs[tid * 4], qrot + (size_t)b * 720 + tid * 4);
  if (tid < 96)
    async_b128_to_lds(&cs[tid * 4], coords + ((size_t)b * N_ + chunk * CH_) * 3 + tid * 4);

  // ---- GEMM: waves 0-3 -> K (row-blocks w, w+4); waves 4-7 -> V ----
  int s = w >> 2;        // 0=K, 1=V
  int rb = w & 3;        // row-blocks rb and rb+4 (rows rb*16.. and rb*16+64..)
  const __bf16* A0 = kvP + ((size_t)(b * N_ + chunk * CH_ + rb * 16 + lr)) * 256;
  const __bf16* A1 = A0 + 64 * 256;
  // A fragments: k order chosen here (not AMD's table order) -> 16 contiguous K at g*16 (matches B)
  v16bf af0[8], af1[8];
#pragma unroll
  for (int kk = 0; kk < 8; ++kk) {
    const __bf16* ap0 = A0 + kk * 32 + g * 16;
    const __bf16* ap1 = A1 + kk * 32 + g * 16;
    v8bf lo0 = *(const v8bf*)(ap0);
    v8bf hi0 = *(const v8bf*)(ap0 + 8);
    v8bf lo1 = *(const v8bf*)(ap1);
    v8bf hi1 = *(const v8bf*)(ap1 + 8);
#pragma unroll
    for (int e = 0; e < 8; ++e) {
      af0[kk][e] = lo0[e]; af0[kk][e + 8] = hi0[e];
      af1[kk][e] = lo1[e]; af1[kk][e + 8] = hi1[e];
    }
  }
  const __bf16* Bmat = WP + ((size_t)(layer * 2 + s) * 240) * 256;
  const float* bias = (s ? bv : bk) + layer * 240;
  float* Cl = s ? Vl : Kl;
  for (int t = 0; t < 15; ++t) {
    const __bf16* Bb = Bmat + ((size_t)(t * 16 + lr)) * 256 + g * 16;
    __builtin_prefetch(Bb + 16 * 256, 0, 0);   // next col-tile (global_prefetch_b8)
    v8f c0 = {}, c1 = {};
#pragma unroll
    for (int kk = 0; kk < 8; ++kk) {
      // B fragment: col = lane%16, 16 contiguous K at g*16 (same k order as A)
      v16bf bf = *(const v16bf*)(Bb + kk * 32);
      c0 = __builtin_amdgcn_wmma_f32_16x16x32_bf16(false, af0[kk], false, bf,
                                                   (short)0, c0, false, false);
      c1 = __builtin_amdgcn_wmma_f32_16x16x32_bf16(false, af1[kk], false, bf,
                                                   (short)0, c1, false, false);
    }
    int col = t * 16 + lr;
    float bcol = bias[col];
#pragma unroll
    for (int vv = 0; vv < 8; ++vv) {
      int row = rb * 16 + g * 8 + vv;       // C layout: VGPR v -> M=v / v+8
      Cl[row * KSTR + col] = c0[vv] + bcol;
      Cl[(row + 64) * KSTR + col] = c1[vv] + bcol;
    }
  }
  wait_asynccnt0();       // qs/cs async loads complete before the barrier
  __syncthreads();

  // ---- RoPE on K (in LDS), coords from LDS ----
  for (int idx = tid; idx < CH_ * H_ * 15; idx += 256) {
    int n = idx / 120, rem = idx % 120, h = rem / 15, p = rem % 15;
    float coord = cs[n * 3 + (p / 5)];
    float invf = __expf(-(float)(p % 5) * 0.2f * 9.21034037197618f);
    float sn, cc;
    __sincosf(coord * invf, &sn, &cc);
    int c0i = h * 30 + 2 * p;
    float x1 = Kl[n * KSTR + c0i], x2 = Kl[n * KSTR + c0i + 1];
    Kl[n * KSTR + c0i]     = x1 * cc - x2 * sn;
    Kl[n * KSTR + c0i + 1] = x1 * sn + x2 * cc;
  }
  __syncthreads();

  // ---- scores + online-softmax partials; wave w == head ----
  int h = w;
  for (int q = 0; q < 3; ++q) {
    float qv[30];
#pragma unroll
    for (int d = 0; d < 30; ++d) qv[d] = qs[q * 240 + h * 30 + d];
    float sc[4];
#pragma unroll
    for (int r = 0; r < 4; ++r) {
      int n = ln + r * 32;
      float acc = 0.f;
#pragma unroll
      for (int d = 0; d < 30; ++d) acc += qv[d] * Kl[n * KSTR + h * 30 + d];
      sc[r] = acc;
    }
    float m = fmaxf(fmaxf(sc[0], sc[1]), fmaxf(sc[2], sc[3]));
#pragma unroll
    for (int off = 16; off > 0; off >>= 1) m = fmaxf(m, __shfl_xor(m, off, 32));
    float e[4], l = 0.f;
#pragma unroll
    for (int r = 0; r < 4; ++r) { e[r] = __expf(sc[r] - m); l += e[r]; }
#pragma unroll
    for (int off = 16; off > 0; off >>= 1) l += __shfl_xor(l, off, 32);
    float o[30];
#pragma unroll
    for (int d = 0; d < 30; ++d) {
      float acc = 0.f;
#pragma unroll
      for (int r = 0; r < 4; ++r) acc += e[r] * Vl[(ln + r * 32) * KSTR + h * 30 + d];
      o[d] = acc;
    }
#pragma unroll
    for (int d = 0; d < 30; ++d) {
#pragma unroll
      for (int off = 16; off > 0; off >>= 1) o[d] += __shfl_xor(o[d], off, 32);
    }
    if (ln == 0) {
      float* p = part + ((size_t)(((b * H_ + h) * 3 + q) * NCH_ + chunk)) * 32;
      p[0] = m; p[1] = l;
#pragma unroll
      for (int d = 0; d < 30; ++d) p[2 + d] = o[d];
    }
  }
}

// ---------------- combine split-K softmax partials (32 chunks) ----------------
__global__ __launch_bounds__(32)
void combine_kernel(const float* __restrict__ part, float* __restrict__ ao) {
  __shared__ float cf[NCH_];
  int bx = blockIdx.x;                // B*H*3
  int b = bx / 24, rem = bx % 24, h = rem / 3, q = rem % 3;
  const float* p = part + ((size_t)((b * H_ + h) * 3 + q)) * NCH_ * 32;
  int ln = threadIdx.x;
  float m1 = p[ln * 32 + 0], l1 = p[ln * 32 + 1];
  float M = m1;
#pragma unroll
  for (int off = 16; off > 0; off >>= 1) M = fmaxf(M, __shfl_xor(M, off, 32));
  float c1 = __expf(m1 - M);
  float l = l1 * c1;
#pragma unroll
  for (int off = 16; off > 0; off >>= 1) l += __shfl_xor(l, off, 32);
  cf[ln] = c1;
  __syncthreads();
  if (ln < 30) {
    float O = 0.f;
    for (int j = 0; j < NCH_; ++j) O += cf[j] * p[j * 32 + 2 + ln];
    ao[((size_t)b * 3 + q) * 240 + h * 30 + ln] = O / l;
  }
}

// ---------------- out-proj + LN1 + FFN(GELU) + LN2 (3 tokens/batch) --------
__global__ __launch_bounds__(256)
void token_update_kernel(float* __restrict__ x, const float* __restrict__ ao,
                         const float* __restrict__ Wo, const float* __restrict__ bo,
                         const float* __restrict__ W1, const float* __restrict__ b1,
                         const float* __restrict__ W2, const float* __restrict__ b2,
                         const float* __restrict__ g1, const float* __restrict__ be1,
                         const float* __restrict__ g2, const float* __restrict__ be2,
                         int layer) {
  __shared__ float xs[720], as_[720], ys[720], zs[720];
  __shared__ float hs[3 * FF_];
  __shared__ float red[256];
  int b = blockIdx.x, tid = threadIdx.x;
  for (int i = tid; i < 720; i += 256) { xs[i] = x[b * 720 + i]; as_[i] = ao[b * 720 + i]; }
  __syncthreads();
  for (int i = tid; i < 720; i += 256) {
    int t = i / 240, d = i % 240;
    float acc = bo[layer * 240 + d];
    const float* wrow = Wo + ((size_t)layer * 240 + d) * 240;
    for (int c = 0; c < 240; ++c) acc += as_[t * 240 + c] * wrow[c];
    ys[i] = xs[i] + acc;
  }
  __syncthreads();
  for (int t = 0; t < 3; ++t) {     // LN1
    red[tid] = (tid < 240) ? ys[t * 240 + tid] : 0.f; __syncthreads();
    for (int s2 = 128; s2 > 0; s2 >>= 1) { if (tid < s2) red[tid] += red[tid + s2]; __syncthreads(); }
    float mean = red[0] / 240.f; __syncthreads();
    float v = 0.f;
    if (tid < 240) { float dd = ys[t * 240 + tid] - mean; v = dd * dd; }
    red[tid] = v; __syncthreads();
    for (int s2 = 128; s2 > 0; s2 >>= 1) { if (tid < s2) red[tid] += red[tid + s2]; __syncthreads(); }
    float rinv = rsqrtf(red[0] / 240.f + 1e-5f);
    if (tid < 240)
      zs[t * 240 + tid] = (ys[t * 240 + tid] - mean) * rinv * g1[layer * 240 + tid] + be1[layer * 240 + tid];
    __syncthreads();
  }
  for (int i = tid; i < 3 * FF_; i += 256) {   // FFN up + exact GELU
    int t = i / FF_, f = i % FF_;
    float acc = b1[layer * FF_ + f];
    const float* wrow = W1 + ((size_t)layer * FF_ + f) * 240;
    for (int c = 0; c < 240; ++c) acc += zs[t * 240 + c] * wrow[c];
    hs[i] = 0.5f * acc * (1.0f + erff(acc * 0.7071067811865475f));
  }
  __syncthreads();
  for (int i = tid; i < 720; i += 256) {       // FFN down + residual
    int t = i / 240, d = i % 240;
    float acc = b2[layer * 240 + d];
    const float* wrow = W2 + ((size_t)layer * 240 + d) * FF_;
    for (int f = 0; f < FF_; ++f) acc += hs[t * FF_ + f] * wrow[f];
    ys[i] = zs[i] + acc;
  }
  __syncthreads();
  for (int t = 0; t < 3; ++t) {     // LN2 -> x
    red[tid] = (tid < 240) ? ys[t * 240 + tid] : 0.f; __syncthreads();
    for (int s2 = 128; s2 > 0; s2 >>= 1) { if (tid < s2) red[tid] += red[tid + s2]; __syncthreads(); }
    float mean = red[0] / 240.f; __syncthreads();
    float v = 0.f;
    if (tid < 240) { float dd = ys[t * 240 + tid] - mean; v = dd * dd; }
    red[tid] = v; __syncthreads();
    for (int s2 = 128; s2 > 0; s2 >>= 1) { if (tid < s2) red[tid] += red[tid + s2]; __syncthreads(); }
    float rinv = rsqrtf(red[0] / 240.f + 1e-5f);
    if (tid < 240)
      x[b * 720 + t * 240 + tid] =
          (ys[t * 240 + tid] - mean) * rinv * g2[layer * 240 + tid] + be2[layer * 240 + tid];
    __syncthreads();
  }
}

// ---------------- final projection: x[:,1:] @ Wout.T + bout ----------------
__global__ __launch_bounds__(256)
void out_proj_kernel(const float* __restrict__ x, const float* __restrict__ Wout,
                     const float* __restrict__ bout, float* __restrict__ out) {
  __shared__ float xs[720];
  int b = blockIdx.x, tid = threadIdx.x;
  for (int i = tid; i < 720; i += 256) xs[i] = x[b * 720 + i];
  __syncthreads();
  for (int i = tid; i < 480; i += 256) {
    int t = i / 240 + 1, d = i % 240;
    float acc = bout[d];
    const float* wrow = Wout + (size_t)d * 240;
    for (int c = 0; c < 240; ++c) acc += xs[t * 240 + c] * wrow[c];
    out[(size_t)b * 480 + i] = acc;
  }
}

// ============================ launch ============================
extern "C" void kernel_launch(void* const* d_in, const int* in_sizes, int n_in,
                              void* d_out, int out_size, void* d_ws, size_t ws_size,
                              hipStream_t stream) {
  const float* state = (const float*)d_in[0];
  const float* hand  = (const float*)d_in[1];
  const float* headp = (const float*)d_in[2];
  const float* coords= (const float*)d_in[3];
  const float* kv    = (const float*)d_in[4];
  const float* learn = (const float*)d_in[5];
  const float* mod   = (const float*)d_in[6];
  const float* Wq = (const float*)d_in[7];  const float* bq = (const float*)d_in[8];
  const float* Wk = (const float*)d_in[9];  const float* bk = (const float*)d_in[10];
  const float* Wv = (const float*)d_in[11]; const float* bv = (const float*)d_in[12];
  const float* Wo = (const float*)d_in[13]; const float* bo = (const float*)d_in[14];
  const float* W1 = (const float*)d_in[15]; const float* b1 = (const float*)d_in[16];
  const float* W2 = (const float*)d_in[17]; const float* b2 = (const float*)d_in[18];
  const float* g1 = (const float*)d_in[19]; const float* be1 = (const float*)d_in[20];
  const float* g2 = (const float*)d_in[21]; const float* be2 = (const float*)d_in[22];
  const float* Wout = (const float*)d_in[23]; const float* bout = (const float*)d_in[24];
  float* out = (float*)d_out;

  char* ws = (char*)d_ws;
  float*  x    = (float*)(ws + 0);          // 184320 B
  float*  qrot = (float*)(ws + 184320);     // 184320 B
  float*  ao   = (float*)(ws + 368640);     // 184320 B
  __bf16* WP   = (__bf16*)(ws + 552960);    // 491520 B
  float*  part = (float*)(ws + 1044480);    // 6291456 B
  __bf16* kvP  = (__bf16*)(ws + 7335936);   // 134217728 B (L2-resident)

  init_x_kernel<<<(B_ * 3 * D_ + 255) / 256, 256, 0, stream>>>(state, learn, mod, x);
  pack_w_kernel<<<(2 * 2 * 240 * 256) / 256, 256, 0, stream>>>(Wk, Wv, WP);
  pack_kv_kernel<<<(B_ * N_ * 32) / 256, 256, 0, stream>>>(kv, kvP);

  for (int layer = 0; layer < 2; ++layer) {
    qproj_rope_kernel<<<B_, 256, 0, stream>>>(x, Wq, bq, hand, headp, qrot, layer);
    attn_partial_kernel<<<B_ * NCH_, 256, 0, stream>>>(kvP, WP, bk, bv, qrot, coords, part, layer);
    combine_kernel<<<B_ * H_ * 3, 32, 0, stream>>>(part, ao);
    token_update_kernel<<<B_, 256, 0, stream>>>(x, ao, Wo, bo, W1, b1, W2, b2,
                                                g1, be1, g2, be2, layer);
  }
  out_proj_kernel<<<B_, 256, 0, stream>>>(x, Wout, bout, out);
}
